// ASPestNet_6021544149329
// MI455X (gfx1250) — matrix operand order repaired and checked by reading, not compiled, not run on hardware
//
// ASPestNet forward for MI455X (gfx1250), compile-only target.
// Dense contractions via v_wmma_f32_16x16x32_f16 (f16 operands, f32 accum).
// CDNA5 paths: WMMA everywhere, TDM (tensor_load_to_lds) double-buffered B-panel
// staging with s_wait_tensorcnt, global_prefetch on the iDFT coef stream.
#include <hip/hip_runtime.h>
#include <hip/hip_bf16.h>
#include <math.h>

typedef __attribute__((ext_vector_type(16))) _Float16 v16h;
typedef __attribute__((ext_vector_type(8)))  _Float16 v8h;
typedef __attribute__((ext_vector_type(8)))  float    v8f;
typedef unsigned int uintv4 __attribute__((ext_vector_type(4)));
typedef int intv4 __attribute__((ext_vector_type(4)));
typedef int intv8 __attribute__((ext_vector_type(8)));

#define DEV static __device__ __forceinline__

#if __has_builtin(__builtin_amdgcn_tensor_load_to_lds)
#define HAS_TDM 1
#else
#define HAS_TDM 0
#endif

// ---------------- WMMA fragment loaders (contiguous k order, same in A and B) ----
union AFrag { v16h v; v8h h[2]; };

// A (16x32 f16): lane<16 holds K=[0..15]; lane>=16 holds K=[16..31]
DEV v16h load_a16(const _Float16* A, int ldk, int row, int kc, int lane){
  AFrag f;
  const _Float16* p = A + (long)row * ldk + kc + ((lane >> 4) << 4);
  f.h[0] = *(const v8h*)p;
  f.h[1] = *(const v8h*)(p + 8);
  return f.v;
}
// B (32x16 f16), stored K-contiguous per column: lane&15 = column, lane>>4 selects K half
DEV v16h load_bt16(const _Float16* Bt, int ldk, int col, int kc, int lane){
  return *(const v16h*)(Bt + (long)col * ldk + kc + ((lane >> 4) << 4));
}
#define WMMA_F16(a,b,c) __builtin_amdgcn_wmma_f32_16x16x32_f16(false,(a),false,(b),(short)0,(c),false,false)

// ---------------- TDM: 2D tile (tileW x tileH halfs) global -> LDS ------------
// D# per CDNA5 ISA: group0 = {count/lds_addr/global_addr/type},
// group1 = {data_size, tensor dims/strides, tile dims}. Groups 2/3 zero (2D).
DEV void tdm_load_2d(unsigned ldsAddr, const void* g, unsigned tdim0, unsigned tdim1,
                     unsigned tileW, unsigned tileH, unsigned strideElems){
#if HAS_TDM
  unsigned long long ga = (unsigned long long)(size_t)g;
  uintv4 g0;
  g0.x = 1u;                                               // count=1 (user D#)
  g0.y = ldsAddr;                                          // LDS byte address
  g0.z = (unsigned)(ga & 0xFFFFFFFFull);                   // global_addr[31:0]
  g0.w = (unsigned)((ga >> 32) & 0x01FFFFFFull) | (2u << 30); // addr[56:32] | type=2
  intv8 g1;
  g1[0] = (int)(1u << 16);                                 // data_size = 2 bytes
  g1[1] = (int)((tdim0 & 0xFFFFu) << 16);                  // tensor_dim0[15:0]
  g1[2] = (int)(((tdim0 >> 16) & 0xFFFFu) | ((tdim1 & 0xFFFFu) << 16));
  g1[3] = (int)(((tdim1 >> 16) & 0xFFFFu) | ((tileW & 0xFFFFu) << 16)); // tile_dim0
  g1[4] = (int)(tileH & 0xFFFFu);                          // tile_dim1 (tile_dim2=0)
  g1[5] = (int)strideElems;                                // tensor_dim0_stride[31:0]
  g1[6] = 0; g1[7] = 0;
  intv4 z4 = {0,0,0,0};
#if defined(__clang_major__) && (__clang_major__ >= 23)
  intv8 z8 = {0,0,0,0,0,0,0,0};
  __builtin_amdgcn_tensor_load_to_lds(g0, g1, z4, z4, z8, 0);
#else
  __builtin_amdgcn_tensor_load_to_lds(g0, g1, z4, z4, 0);
#endif
#endif
}

// ---------------- small math helpers ----------------
struct c32 { float x, y; };
DEV c32 cmul(c32 a, c32 b){ return { a.x*b.x - a.y*b.y, a.x*b.y + a.y*b.x }; }
DEV c32 cdiv(c32 a, c32 b){
  float d = 1.f / (b.x*b.x + b.y*b.y);
  return { (a.x*b.x + a.y*b.y)*d, (a.y*b.x - a.x*b.y)*d };
}
DEV c32 cinv(c32 b){ float d = 1.f/(b.x*b.x + b.y*b.y); return { b.x*d, -b.y*d }; }
DEV float sigm(float x){ return 1.f/(1.f + expf(-x)); }
DEV float sp2f(float x){ float sp = (x > 20.f) ? x : log1pf(expf(x)); return sp * 1.4426950408889634f; }
DEV float ftanf_(float x){ return tanf(1.5707963267948966f * sigm(x)); }

__constant__ float D_FDN_c[6] = {233.f,311.f,421.f,461.f,587.f,613.f};
__constant__ float D_AP_c[6][4] = {
  {131.f,151.f,337.f,353.f},{103.f,173.f,331.f,373.f},{89.f,181.f,307.f,401.f},
  {79.f,197.f,281.f,419.f},{61.f,211.f,257.f,431.f},{47.f,229.f,251.f,443.f}};

// ================= constants =================
#define SR_     48000
#define NFFT_   1024
#define HOP_    128
#define NBINS_  513
#define NBP_    528
#define BB_     8
#define LL_     85504
#define T0_     669
#define ROWS_   5352
#define ROWSP_  5360
#define NF_     43201
#define NFP_    43232
#define IRLEN_  86400
#define TP_     76
#define QDIAG_  0.6666666666666667f

// ================= kernels =================

__global__ void k_basis(_Float16* Bc, _Float16* Bs){
  int idx = blockIdx.x * blockDim.x + threadIdx.x;
  if (idx >= NBP_ * NFFT_) return;
  int f = idx >> 10, n = idx & 1023;
  float c = 0.f, s = 0.f;
  if (f < NBINS_){
    float fr  = 20.0f * powf(600.0f, (float)f / 512.0f);
    float w   = 0.5f - 0.5f * cosf(6.283185307179586f * (float)n / 1023.0f);
    float ang = 6.283185307179586f * fr * (float)n / 48000.0f;
    c = w * cosf(ang); s = w * sinf(ang);
  }
  Bc[idx] = (_Float16)c; Bs[idx] = (_Float16)s;
}

__global__ void k_frames(const float* x, _Float16* F){
  int idx = blockIdx.x * blockDim.x + threadIdx.x;
  if (idx >= ROWSP_ * NFFT_) return;
  int row = idx >> 10, n = idx & 1023;
  float v = 0.f;
  if (row < ROWS_){
    int b = row / T0_, t = row % T0_;
    int src = t * HOP_ + n - (NFFT_/2);
    if (src < 0) src = -src;
    if (src >= LL_) src = 2*LL_ - 2 - src;
    v = x[(long)b * LL_ + src];
  }
  F[idx] = (_Float16)v;
}

// STFT log-magnitude: 2 WMMA accumulators (cos & sin basis), store (B,513,669) f16
__global__ void k_stft(const _Float16* F, const _Float16* Bc, const _Float16* Bs,
                       _Float16* spec){
  int lane = threadIdx.x;
  int tm = blockIdx.x, tn = blockIdx.y;
  int row = tm*16 + (lane & 15);
  int col = tn*16 + (lane & 15);
  v8f accR = {}, accI = {};
  for (int kc = 0; kc < NFFT_; kc += 32){
    v16h a  = load_a16(F, NFFT_, row, kc, lane);
    v16h bc = load_bt16(Bc, NFFT_, col, kc, lane);
    v16h bs = load_bt16(Bs, NFFT_, col, kc, lane);
    accR = WMMA_F16(a, bc, accR);
    accI = WMMA_F16(a, bs, accI);
  }
  int rowBase = tm*16 + ((lane >> 4) << 3);
  if (col < NBINS_){
#pragma unroll
    for (int r = 0; r < 8; ++r){
      int rr = rowBase + r;
      if (rr < ROWS_){
        float re = accR[r], im = accI[r];
        float v = logf(sqrtf(re*re + im*im) + 1e-7f);
        int b = rr / T0_, t = rr % T0_;
        spec[((long)b * NBINS_ + col) * T0_ + t] = (_Float16)v;
      }
    }
  }
}

// f32 (N x K) -> f16 (N x Kp) zero-padded (identity K order)
__global__ void k_w16(const float* W, _Float16* Wh, int N, int K, int Kp){
  int idx = blockIdx.x * blockDim.x + threadIdx.x;
  if (idx >= N * Kp) return;
  int n = idx / Kp, k = idx - n * Kp;
  Wh[idx] = (_Float16)(k < K ? W[(long)n * K + k] : 0.f);
}

// conv weight f32 OIHW -> f16 with channel-fastest K order: k = (kh*KW+kw)*Cin + cin
__global__ void k_w16c(const float* W, _Float16* Wh, int N, int K, int Kp,
                       int Cin, int KHW){
  int idx = blockIdx.x * blockDim.x + threadIdx.x;
  if (idx >= N * Kp) return;
  int n = idx / Kp, kk = idx - n * Kp;
  float v = 0.f;
  if (kk < K){
    int khw = kk / Cin, cin = kk - khw * Cin;
    v = W[(long)n * K + cin * KHW + khw];
  }
  Wh[idx] = (_Float16)v;
}

// Implicit-GEMM conv, compile-time geometry. K order is channel-fastest, so for
// CIN>=16 each 16-wide B fragment has constant (kh,kw) and cin = base + j
// (constant-stride gather, no divisions in the hot loop).
template<int CIN, int KH, int KW, int SH, int SW, int KREAL, int KP>
__global__ void k_convT(const _Float16* in, const _Float16* Wh, const float* bias,
                        _Float16* out, int IH, int IW, int Cout, int OH, int OW){
  int lane = threadIdx.x;
  int tm = blockIdx.x, tn = blockIdx.y, b = blockIdx.z;
  int HW = OH * OW;
  int pos = tn*16 + (lane & 15);
  bool valid = pos < HW;
  int oh = valid ? pos / OW : 0;
  int ow = valid ? pos - oh * OW : 0;
  int wrow = tm*16 + (lane & 15);
  const long CHW = (long)IH * IW;
  const _Float16* inb = in + (long)b * CIN * CHW + (long)(oh * SH) * IW + ow * SW;
  int kk0 = (lane >> 4) << 4;
  v8f acc = {};
  for (int kc = 0; kc < KP; kc += 32){
    v16h a = load_a16(Wh, KP, wrow, kc, lane);
    int k0 = kc + kk0;
    v16h bf;
    if (CIN >= 16){
      int cin0 = k0 & (CIN - 1);
      int khw  = k0 / CIN;                 // constant shift
      int kh = khw / KW, kw = khw - kh * KW; // constant divisors
      const _Float16* p = inb + (long)cin0 * CHW + kh * IW + kw;
#pragma unroll
      for (int j = 0; j < 16; ++j){
        float v = valid ? (float)p[(long)j * CHW] : 0.f;
        bf[j] = (_Float16)v;
      }
    } else {
#pragma unroll
      for (int j = 0; j < 16; ++j){
        int k = k0 + j;
        float v = 0.f;
        if (valid && k < KREAL){
          int kh = k / KW, kw = k - kh * KW;  // CIN==1: k == khw
          v = (float)inb[kh * IW + kw];
        }
        bf[j] = (_Float16)v;
      }
    }
    acc = WMMA_F16(a, bf, acc);
  }
  if (valid){
    int rowBase = tm*16 + ((lane >> 4) << 3);
#pragma unroll
    for (int r = 0; r < 8; ++r){
      int co = rowBase + r;
      float v = acc[r] + bias[co];
      v = v > 0.f ? v : 0.f;
      out[((long)b * Cout + co) * HW + pos] = (_Float16)v;
    }
  }
}

// Generic GEMM (B from global): out = A(M x Kp) @ W(N x Kp)^T + bias
__global__ void k_gemm(const _Float16* A, const _Float16* Wt, const float* bias,
                       float* outF, _Float16* outH, int Kp, int N, int relu){
  int lane = threadIdx.x;
  int tm = blockIdx.x, tn = blockIdx.y;
  int row = tm*16 + (lane & 15);
  int col = tn*16 + (lane & 15);
  v8f acc = {};
  for (int kc = 0; kc < Kp; kc += 32){
    v16h a = load_a16(A, Kp, row, kc, lane);
    v16h w = load_bt16(Wt, Kp, col, kc, lane);
    acc = WMMA_F16(a, w, acc);
  }
  int rowBase = tm*16 + ((lane >> 4) << 3);
  float bcv = bias ? bias[col] : 0.f;
#pragma unroll
  for (int r = 0; r < 8; ++r){
    int rr = rowBase + r;
    float v = acc[r] + bcv;
    if (relu && v < 0.f) v = 0.f;
    long o = (long)rr * N + col;
    if (outF) outF[o] = v;
    if (outH) outH[o] = (_Float16)v;
  }
}

// GEMM with TDM-staged B panel: 16x32-half tiles DMA'd to LDS, double buffered,
// tracked with TENSORcnt (s_wait_tensorcnt). One wave per block.
__global__ void k_gemm_tdm(const _Float16* A, const _Float16* Wt, const float* bias,
                           float* outF, _Float16* outH, int Kp, int N, int relu){
  int lane = threadIdx.x;
  int tm = blockIdx.x, tn = blockIdx.y;
  int row = tm*16 + (lane & 15);
  int col = tn*16 + (lane & 15);
  v8f acc = {};
#if HAS_TDM
  __shared__ __attribute__((aligned(32))) _Float16 bstage[2][16*32];
  const _Float16* wpanel = Wt + (long)tn * 16 * Kp;
  unsigned lds0 = (unsigned)(size_t)&bstage[0][0];
  unsigned lds1 = (unsigned)(size_t)&bstage[1][0];
  tdm_load_2d(lds0, wpanel, 32, 16, 32, 16, (unsigned)Kp);
  int buf = 0;
  for (int kc = 0; kc < Kp; kc += 32){
    if (kc + 32 < Kp){
      tdm_load_2d(buf ? lds0 : lds1, wpanel + kc + 32, 32, 16, 32, 16, (unsigned)Kp);
      __builtin_amdgcn_s_wait_tensorcnt(1);   // oldest (current buf) complete
    } else {
      __builtin_amdgcn_s_wait_tensorcnt(0);
    }
    __syncthreads();
    v16h b = *(const v16h*)(&bstage[buf][(lane & 15)*32 + ((lane >> 4) << 4)]);
    v16h a = load_a16(A, Kp, row, kc, lane);
    acc = WMMA_F16(a, b, acc);
    __syncthreads();
    buf ^= 1;
  }
#else
  for (int kc = 0; kc < Kp; kc += 32){
    v16h a = load_a16(A, Kp, row, kc, lane);
    v16h w = load_bt16(Wt, Kp, col, kc, lane);
    acc = WMMA_F16(a, w, acc);
  }
#endif
  int rowBase = tm*16 + ((lane >> 4) << 3);
  float bcv = bias ? bias[col] : 0.f;
#pragma unroll
  for (int r = 0; r < 8; ++r){
    int rr = rowBase + r;
    float v = acc[r] + bcv;
    if (relu && v < 0.f) v = 0.f;
    long o = (long)rr * N + col;
    if (outF) outF[o] = v;
    if (outH) outH[o] = (_Float16)v;
  }
}

// conv5 (B,128,56,76) NCHW f16 -> sequence buffer ((b*76+t)*56+fd, 128)
__global__ void k_seq(const _Float16* c5, _Float16* seq){
  int idx = blockIdx.x * blockDim.x + threadIdx.x;
  if (idx >= 34048 * 128) return;
  int row = idx >> 7, c = idx & 127;
  int bt = row / 56, fd = row - bt * 56;
  int b = bt / TP_, t = bt - b * TP_;
  seq[idx] = c5[(((long)b * 128 + c) * 56 + fd) * TP_ + t];
}

// One GRU direction scan; xg f32 (S,T,3H) precomputed by GEMM; block = one sequence.
__global__ void k_gru(const float* xg, const float* U, const float* bh,
                      _Float16* y, int T, int H, int backward,
                      int outStride, int outOff){
  extern __shared__ float smem[];
  float* hbuf = smem;
  float* gh   = smem + H;
  int s = blockIdx.x;
  int j = threadIdx.x;
  if (j < H) hbuf[j] = 0.f;
  __syncthreads();
  const float* xgs = xg + (long)s * T * 3 * H;
  for (int t = 0; t < T; ++t){
    int tt = backward ? (T - 1 - t) : t;
    float g = bh[j];
    for (int k = 0; k < H; ++k) g += U[(long)j * H + k] * hbuf[k];
    gh[j] = g;
    __syncthreads();
    float hn = 0.f;
    if (j < H){
      const float* xt = xgs + (long)tt * 3 * H;
      float r = sigm(xt[j]        + gh[j]);
      float z = sigm(xt[H + j]    + gh[H + j]);
      float n = tanhf(xt[2*H + j] + r * gh[2*H + j]);
      hn = (1.f - z) * n + z * hbuf[j];
    }
    __syncthreads();
    if (j < H){
      hbuf[j] = hn;
      y[((long)s * T + tt) * outStride + outOff + j] = (_Float16)hn;
    }
    __syncthreads();
  }
}

// LayerNorm(256) + affine + ReLU
__global__ void k_ln(const float* X, const float* g, const float* be,
                     float* outF, _Float16* outH){
  __shared__ float red[256];
  int row = blockIdx.x, j = threadIdx.x;
  float v = X[(long)row * 256 + j];
  red[j] = v; __syncthreads();
  for (int s = 128; s > 0; s >>= 1){ if (j < s) red[j] += red[j + s]; __syncthreads(); }
  float mu = red[0] * (1.f/256.f); __syncthreads();
  float d = v - mu;
  red[j] = d * d; __syncthreads();
  for (int s = 128; s > 0; s >>= 1){ if (j < s) red[j] += red[j + s]; __syncthreads(); }
  float var = red[0] * (1.f/256.f);
  float o = d * rsqrtf(var + 1e-5f) * g[j] + be[j];
  if (o < 0.f) o = 0.f;
  outF[(long)row * 256 + j] = o;
  outH[(long)row * 256 + j] = (_Float16)o;
}

// _proj: out[b][i][j] = sum_c (sum_t enc[b,t,c] W1[i,t] + b1[i]) * W2[j,c] + b2[j]
__global__ void k_proj(const float* enc, const float* W1, const float* b1,
                       const float* W2, const float* b2, float* out,
                       int z1, int z2){
  __shared__ float v[256];
  int b = blockIdx.x / z1, i = blockIdx.x - b * z1;
  int c = threadIdx.x;
  float a = 0.f;
  for (int t = 0; t < TP_; ++t) a += enc[((long)b * TP_ + t) * 256 + c] * W1[i * TP_ + t];
  v[c] = a + b1[i];
  __syncthreads();
  if (c < z2){
    float o = b2[c];
    for (int k = 0; k < 256; ++k) o += v[k] * W2[c * 256 + k];
    out[((long)b * z1 + i) * z2 + c] = o;
  }
}

DEV c32 svf1(c32 zi, c32 zi2, float f, float R, float mLP, float mBP, float mHP){
  float f2 = f * f;
  float n0 = f2*mLP + f*mBP + mHP;
  float n1 = 2.f*f2*mLP - 2.f*mHP;
  float n2 = f2*mLP - f*mBP + mHP;
  float d0 = f2 + 2.f*R*f + 1.f;
  float d1 = 2.f*f2 - 2.f;
  float d2 = f2 - 2.f*R*f + 1.f;
  c32 num = { n0 + n1*zi.x + n2*zi2.x, n1*zi.y + n2*zi2.y };
  c32 den = { d0 + d1*zi.x + d2*zi2.x, d1*zi.y + d2*zi2.y };
  return cdiv(num, den);
}

__global__ void k_freq(const float* za, const float* bcP, const float* fC1P,
                       const float* RC1P, const float* mC1P, const float* fCdP,
                       const float* GCdP, const float* RCdP, const float* SAP,
                       float* HRe, float* HIm){
  int f = blockIdx.x * blockDim.x + threadIdx.x;
  int b = blockIdx.y;
  if (f >= NF_) return;
  float th = za[f];
  c32 zi  = { cosf(th), -sinf(th) };
  c32 zi2 = cmul(zi, zi);
  c32 C1 = {1.f, 0.f};
#pragma unroll
  for (int j = 0; j < 8; ++j){
    float fj = ftanf_(fC1P[b*8 + j]);
    float Rj = sp2f(RC1P[b*8 + j]);
    C1 = cmul(C1, svf1(zi, zi2, fj, Rj,
                       mC1P[(b*3+0)*8+j], mC1P[(b*3+1)*8+j], mC1P[(b*3+2)*8+j]));
  }
  c32 Cd = {1.f, 0.f};
#pragma unroll
  for (int j = 0; j < 8; ++j){
    float fj = ftanf_(fCdP[b*8 + j]);
    float G  = powf(10.f, -sp2f(GCdP[b*8 + j]));
    float Rj = sp2f(RCdP[b*8 + j]);
    float sG = sqrtf(G);
    float mLP = (j == 0) ? G : 1.f;
    float mHP = (j == 7) ? G : 1.f;
    float mBP = 2.f * Rj * ((j == 0 || j == 7) ? sG : G);
    Cd = cmul(Cd, svf1(zi, zi2, fj, Rj, mLP, mBP, mHP));
  }
  c32 Hs = {0.f, 0.f};
#pragma unroll
  for (int k = 0; k < 6; ++k){
    c32 U = {1.f, 0.f};
#pragma unroll
    for (int j = 0; j < 4; ++j){
      float g = sigm(SAP[(b*6 + k)*4 + j]);
      float ang = th * D_AP_c[k][j];
      c32 zd  = { cosf(ang), -sinf(ang) };
      c32 num = { g + zd.x, zd.y };
      c32 den = { 1.f + g*zd.x, g*zd.y };
      U = cmul(U, cdiv(num, den));
    }
    float angD = th * D_FDN_c[k];
    c32 zD = { cosf(angD), sinf(angD) };
    c32 UC = cmul(U, Cd);
    c32 den = { zD.x - QDIAG_*UC.x + 1e-16f, zD.y - QDIAG_*UC.y };
    float w = bcP[(b*2 + 0)*6 + k] * bcP[(b*2 + 1)*6 + k];
    c32 inv = cinv(den);
    Hs.x += w * inv.x; Hs.y += w * inv.y;
  }
  c32 H = cmul(C1, Hs);
  HRe[(long)b * NF_ + f] = H.x;
  HIm[(long)b * NF_ + f] = H.y;
}

// Pack ortho-iDFT coefficients: rows 0-7 = w_k*Re(H_b), rows 8-15 = w_k*Im(H_b)
__global__ void k_dftcoef(const float* HRe, const float* HIm, _Float16* Ac){
  int idx = blockIdx.x * blockDim.x + threadIdx.x;
  if (idx >= 16 * NFP_) return;
  int row = idx / NFP_, k = idx - row * NFP_;
  float v = 0.f;
  if (k < NF_){
    float inv = rsqrtf((float)IRLEN_);
    float wk = (k == 0 || k == NF_ - 1) ? inv : 2.f * inv;
    v = wk * ((row < 8) ? HRe[(long)row * NF_ + k] : HIm[(long)(row - 8) * NF_ + k]);
  }
  Ac[idx] = (_Float16)v;
}

// iDFT via WMMA, basis generated in-register by rotation recurrence (4 FMA/elem),
// exact integer-phase reseed every 1024 k-steps.
__global__ void k_dft(const _Float16* Ac, float* irlate){
  int lane = threadIdx.x;
  int tn = blockIdx.x;
  int col = tn*16 + (lane & 15);
  int phase = (lane >> 4) << 4;
  const float w = 6.283185307179586f / (float)IRLEN_;
  float cd  = cosf(w * (float)col),        sd  = sinf(w * (float)col);
  float c16 = cosf(w * (float)col * 16.f), s16 = sinf(w * (float)col * 16.f);
  long m0 = ((long)col * phase) % IRLEN_;
  float c = cosf(w * (float)m0), s = sinf(w * (float)m0);
  v8f accC = {}, accS = {};
  for (int kc = 0; kc < NFP_; kc += 32){
    __builtin_prefetch(Ac + (long)(lane & 15) * NFP_ + kc + 512, 0, 0);
    if (kc && (((kc >> 5) & 31) == 0)){
      long m = ((long)col * (long)(kc + phase)) % IRLEN_;
      c = cosf(w * (float)m); s = sinf(w * (float)m);
    }
    v16h bc, bs;
#pragma unroll
    for (int j = 0; j < 16; ++j){
      bc[j] = (_Float16)c; bs[j] = (_Float16)s;
      float cn = c*cd - s*sd;
      s = s*cd + c*sd; c = cn;
    }
    { float cn = c*c16 - s*s16; s = s*c16 + c*s16; c = cn; }
    v16h a = load_a16(Ac, NFP_, lane & 15, kc, lane);
    accC = WMMA_F16(a, bc, accC);
    accS = WMMA_F16(a, bs, accS);
  }
#pragma unroll
  for (int r = 0; r < 8; ++r){
    float other = __shfl_xor(accS[r], 16, 32);
    if (lane < 16) irlate[(long)r * IRLEN_ + col] = accC[r] - other;
  }
}

// Assemble outputs: d_out = [ir | ir_late | h0p], each (8, 86400)
__global__ void k_out(const float* irlate, const float* h0P, float* out){
  int idx = blockIdx.x * blockDim.x + threadIdx.x;
  if (idx >= BB_ * IRLEN_) return;
  int b = idx / IRLEN_, n = idx - b * IRLEN_;
  float late = irlate[idx];
  float h0 = (n < 232) ? h0P[(long)b * 232 + n] : 0.f;
  out[idx] = h0 + late;
  out[(long)BB_*IRLEN_ + idx] = late;
  out[2L*BB_*IRLEN_ + idx] = h0;
}

// ================= host-side orchestration =================
extern "C" void kernel_launch(void* const* d_in, const int* in_sizes, int n_in,
                              void* d_out, int out_size, void* d_ws, size_t ws_size,
                              hipStream_t stream){
  (void)in_sizes; (void)n_in; (void)out_size; (void)ws_size;
  // Input order: x, z_angle, then params flattened in pytree order (sorted dict keys):
  // GCdelta(2-5) RC1(6-9) RCdelta(10-13) SA(14-17) bc(18-21) convW(22-26) convb(27-31)
  // fC1(32-35) fCdelta(36-39) gru1(40-55) gru2(56-63) h(64-67) lin(68-75) mC1(76-79)
  const float* X  = (const float*)d_in[0];
  const float* ZA = (const float*)d_in[1];
  #define PF(i) ((const float*)d_in[(i)])
  const float* convW[5]; const float* convb[5];
  for (int i = 0; i < 5; ++i){ convW[i] = PF(22+i); convb[i] = PF(27+i); }

  size_t off = 0;
  auto alloc = [&](size_t bytes) -> char* {
    char* p = (char*)d_ws + off;
    off += (bytes + 255) & ~(size_t)255;
    return p;
  };
  _Float16* bCos   = (_Float16*)alloc((size_t)NBP_ * NFFT_ * 2);
  _Float16* bSin   = (_Float16*)alloc((size_t)NBP_ * NFFT_ * 2);
  _Float16* frames = (_Float16*)alloc((size_t)ROWSP_ * NFFT_ * 2);
  _Float16* spec   = (_Float16*)alloc((size_t)BB_ * NBINS_ * T0_ * 2);
  _Float16* wbuf   = (_Float16*)alloc((size_t)384 * 7168 * 2);
  _Float16* actA   = (_Float16*)alloc((size_t)BB_ * 64  * 507 * 333 * 2);
  _Float16* actB   = (_Float16*)alloc((size_t)BB_ * 128 * 252 * 329 * 2);
  _Float16* seq    = (_Float16*)alloc((size_t)34048 * 128 * 2);
  float*    xgF    = (float*)   alloc((size_t)34048 * 192 * 4);
  _Float16* y0     = (_Float16*)alloc((size_t)34048 * 128 * 2);
  _Float16* y1     = (_Float16*)alloc((size_t)34048 * 128 * 2);
  _Float16* y2     = (_Float16*)alloc((size_t)608 * 256 * 2);
  float*    linF   = (float*)   alloc((size_t)608 * 256 * 4);
  float*    lnF    = (float*)   alloc((size_t)608 * 256 * 4);
  _Float16* l1h    = (_Float16*)alloc((size_t)608 * 256 * 2);
  float*    encF   = (float*)   alloc((size_t)608 * 256 * 4);
  _Float16* ench   = (_Float16*)alloc((size_t)608 * 256 * 2);
  float*    bcP    = (float*)alloc(8*2*6*4);
  float*    h0P    = (float*)alloc(8*232*4);
  float*    fC1P   = (float*)alloc(8*8*4);
  float*    RC1P   = (float*)alloc(8*8*4);
  float*    mC1P   = (float*)alloc(8*3*8*4);
  float*    fCdP   = (float*)alloc(8*8*4);
  float*    GCdP   = (float*)alloc(8*8*4);
  float*    RCdP   = (float*)alloc(8*8*4);
  float*    SAP    = (float*)alloc(8*6*4*4);
  float*    HRe    = (float*)alloc((size_t)BB_ * NF_ * 4);
  float*    HIm    = (float*)alloc((size_t)BB_ * NF_ * 4);
  _Float16* Ac     = (_Float16*)alloc((size_t)16 * NFP_ * 2);
  float*    irl    = (float*)alloc((size_t)BB_ * IRLEN_ * 4);

  // -------- STFT --------
  k_basis <<<(NBP_*NFFT_ + 255)/256, 256, 0, stream>>>(bCos, bSin);
  k_frames<<<(ROWSP_*NFFT_ + 255)/256, 256, 0, stream>>>(X, frames);
  k_stft  <<<dim3(ROWSP_/16, NBP_/16), dim3(32), 0, stream>>>(frames, bCos, bSin, spec);

  // -------- conv stack (implicit-GEMM WMMA, channel-fastest K) --------
  k_w16c<<<(64*64 + 255)/256, 256, 0, stream>>>(convW[0], wbuf, 64, 35, 64, 1, 35);
  k_convT<1,7,5,1,2,35,64><<<dim3(4,(507*333+15)/16,BB_), dim3(32), 0, stream>>>(
      spec, wbuf, convb[0], actA, 513, 669, 64, 507, 333);
  k_w16c<<<(128*1600 + 255)/256, 256, 0, stream>>>(convW[1], wbuf, 128, 1600, 1600, 64, 25);
  k_convT<64,5,5,2,1,1600,1600><<<dim3(8,(252*329+15)/16,BB_), dim3(32), 0, stream>>>(
      actA, wbuf, convb[1], actB, 507, 333, 128, 252, 329);
  k_w16c<<<(128*3200 + 255)/256, 256, 0, stream>>>(convW[2], wbuf, 128, 3200, 3200, 128, 25);
  k_convT<128,5,5,2,2,3200,3200><<<dim3(8,(124*163+15)/16,BB_), dim3(32), 0, stream>>>(
      actB, wbuf, convb[2], actA, 252, 329, 128, 124, 163);
  k_w16c<<<(128*3200 + 255)/256, 256, 0, stream>>>(convW[3], wbuf, 128, 3200, 3200, 128, 25);
  k_convT<128,5,5,2,2,3200,3200><<<dim3(8,(60*80+15)/16,BB_), dim3(32), 0, stream>>>(
      actA, wbuf, convb[3], actB, 124, 163, 128, 60, 80);
  k_w16c<<<(128*3200 + 255)/256, 256, 0, stream>>>(convW[4], wbuf, 128, 3200, 3200, 128, 25);
  k_convT<128,5,5,1,1,3200,3200><<<dim3(8,(56*76+15)/16,BB_), dim3(32), 0, stream>>>(
      actB, wbuf, convb[4], actA, 60, 80, 128, 56, 76);

  // -------- GRU1 (2 bidirectional layers: S=608, T=56, H=64) --------
  k_seq<<<(34048*128 + 255)/256, 256, 0, stream>>>(actA, seq);
  const _Float16* gin = seq;
  for (int layer = 0; layer < 2; ++layer){
    _Float16* gout = (layer == 0) ? y0 : y1;
    for (int dir = 0; dir < 2; ++dir){
      int base = 40 + (layer*2 + dir)*4;  // W, U, bi, bh
      k_w16<<<(192*128 + 255)/256, 256, 0, stream>>>(PF(base), wbuf, 192, 128, 128);
      k_gemm<<<dim3(34048/16, 192/16), dim3(32), 0, stream>>>(
          gin, wbuf, PF(base+2), xgF, nullptr, 128, 192, 0);
      k_gru<<<608, 192, (64 + 192)*sizeof(float), stream>>>(
          xgF, PF(base+1), PF(base+3), gout, 56, 64, dir, 128, dir*64);
    }
    gin = gout;
  }

  // -------- GRU2 (S=8, T=76, H=128, K=7168) — TDM-staged B panel --------
  for (int dir = 0; dir < 2; ++dir){
    int base = 56 + dir*4;
    k_w16<<<(384*7168 + 255)/256, 256, 0, stream>>>(PF(base), wbuf, 384, 7168, 7168);
    k_gemm_tdm<<<dim3(608/16, 384/16), dim3(32), 0, stream>>>(
        y1, wbuf, PF(base+2), xgF, nullptr, 7168, 384, 0);
    k_gru<<<8, 384, (128 + 384)*sizeof(float), stream>>>(
        xgF, PF(base+1), PF(base+3), y2, TP_, 128, dir, 256, dir*128);
  }

  // -------- linear + LN + ReLU x2 — TDM-staged B panel --------
  const _Float16* lIn = y2;
  for (int l = 0; l < 2; ++l){
    int base = 68 + l*4;  // W, b, g, be
    k_w16<<<(256*256 + 255)/256, 256, 0, stream>>>(PF(base), wbuf, 256, 256, 256);
    k_gemm_tdm<<<dim3(608/16, 256/16), dim3(32), 0, stream>>>(
        lIn, wbuf, PF(base+1), linF, nullptr, 256, 256, 0);
    k_ln<<<608, 256, 0, stream>>>(linF, PF(base+2), PF(base+3),
                                  (l == 0) ? lnF : encF, (l == 0) ? l1h : ench);
    lIn = l1h;
  }

  // -------- projections --------
  k_proj<<<8*2, 256, 0, stream>>>(encF, PF(18), PF(19), PF(20), PF(21), bcP, 2, 6);
  k_proj<<<8*1, 256, 0, stream>>>(encF, PF(64), PF(65), PF(66), PF(67), h0P, 1, 232);
  k_proj<<<8*1, 256, 0, stream>>>(encF, PF(32), PF(33), PF(34), PF(35), fC1P, 1, 8);
  k_proj<<<8*1, 256, 0, stream>>>(encF, PF(6),  PF(7),  PF(8),  PF(9),  RC1P, 1, 8);
  k_proj<<<8*3, 256, 0, stream>>>(encF, PF(76), PF(77), PF(78), PF(79), mC1P, 3, 8);
  k_proj<<<8*1, 256, 0, stream>>>(encF, PF(36), PF(37), PF(38), PF(39), fCdP, 1, 8);
  k_proj<<<8*1, 256, 0, stream>>>(encF, PF(2),  PF(3),  PF(4),  PF(5),  GCdP, 1, 8);
  k_proj<<<8*1, 256, 0, stream>>>(encF, PF(10), PF(11), PF(12), PF(13), RCdP, 1, 8);
  k_proj<<<8*6, 256, 0, stream>>>(encF, PF(14), PF(15), PF(16), PF(17), SAP, 6, 4);

  // -------- frequency-domain synthesis + ortho-iDFT (WMMA) --------
  k_freq<<<dim3((NF_ + 255)/256, BB_), dim3(256), 0, stream>>>(
      ZA, bcP, fC1P, RC1P, mC1P, fCdP, GCdP, RCdP, SAP, HRe, HIm);
  k_dftcoef<<<(16*NFP_ + 255)/256, 256, 0, stream>>>(HRe, HIm, Ac);
  k_dft<<<IRLEN_/16, 32, 0, stream>>>(Ac, irl);
  k_out<<<(BB_*IRLEN_ + 255)/256, 256, 0, stream>>>(irl, h0P, (float*)d_out);
  #undef PF
}
